// MultiHeadAttention_3762391351342
// MI455X (gfx1250) — matrix operand order repaired and checked by reading, not compiled, not run on hardware
//
#include <hip/hip_runtime.h>

typedef _Float16 half_t;
typedef __attribute__((ext_vector_type(8)))  _Float16 v8h;
typedef __attribute__((ext_vector_type(16))) _Float16 v16h;
typedef __attribute__((ext_vector_type(4)))  float    v4f;
typedef __attribute__((ext_vector_type(8)))  float    v8f;

constexpr int    Bn = 8, Sn = 1024, Dn = 1024, Hn = 16, DKn = 64;
constexpr size_t NE = (size_t)Bn * Sn * Dn;   // 8388608 elements per [B,S,D] tensor

static __device__ __forceinline__ v16h cat8(v8h lo, v8h hi) {
  return __builtin_shufflevector(lo, hi, 0,1,2,3,4,5,6,7,8,9,10,11,12,13,14,15);
}
static __device__ __forceinline__ v8f wmma16(v16h a, v16h b, v8f c) {
  // D = A(16x32 f16) * B(32x16 f16) + C(16x16 f32)
  return __builtin_amdgcn_wmma_f32_16x16x32_f16(false, a, false, b, (short)0, c,
                                                false, false);
}

// ---------------- f32 -> f16 weight cast ----------------
__global__ __launch_bounds__(256)
void cvt_kernel(const float* __restrict__ src, half_t* __restrict__ dst) {
  int i = (blockIdx.x * 256 + threadIdx.x) * 4;
  v4f x = *(const v4f*)(src + i);
  half_t* p = dst + i;
  p[0] = (half_t)x[0]; p[1] = (half_t)x[1]; p[2] = (half_t)x[2]; p[3] = (half_t)x[3];
}

// ---------------- generic projection GEMM ----------------
// Y[M=8192, N=1024] = X[M,1024] @ W[1024,1024] + bias
// Block = 128 threads (4 waves), computes 64x64 output tile; wave w does rows w*16..+15.
// XHALF:  X is f16 (ctx) vs f32 (query/value, converted while staging)
// OUTMODE: 0 = f16 row-major [m][n], 1 = f16 V-transposed [(b*1024+n)][s], 2 = f32 row-major + used for d_out
template<bool XHALF, int OUTMODE>
__global__ __launch_bounds__(128)
void gemm_proj(const void* __restrict__ Xv, const half_t* __restrict__ W,
               const float* __restrict__ bias, void* __restrict__ Yv) {
  __shared__ half_t Xs [64][40];   // [row][k]  (pad 40 to dodge bank conflicts)
  __shared__ half_t Wts[64][40];   // [n][k]    transposed W tile

  const int t    = threadIdx.x;
  const int lane = t & 31;
  const int wave = t >> 5;
  const int m0   = blockIdx.x * 64;
  const int n0   = blockIdx.y * 64;

  const int r  = lane & 15;
  const int k0 = (lane < 16) ? 0 : 8;    // A-fragment K base (per §7.12.2)
  const int kh = (lane < 16) ? 0 : 8;    // B-fragment K base

  v8f acc[4] = {};

  for (int kc = 0; kc < Dn / 32; ++kc) {
    __syncthreads();
    // ---- stage X tile 64x32 (as f16) ----
    if (XHALF) {
      const half_t* X = (const half_t*)Xv;
      for (int i = t; i < 256; i += 128) {
        int row = i >> 2, c8 = i & 3;
        *(v8h*)&Xs[row][c8 * 8] =
            *(const v8h*)(X + (size_t)(m0 + row) * Dn + kc * 32 + c8 * 8);
      }
    } else {
      const float* X = (const float*)Xv;
      for (int i = t; i < 512; i += 128) {
        int row = i >> 3, c4 = i & 7;
        v4f x = *(const v4f*)(X + (size_t)(m0 + row) * Dn + kc * 32 + c4 * 4);
        half_t* p = &Xs[row][c4 * 4];
        p[0] = (half_t)x[0]; p[1] = (half_t)x[1];
        p[2] = (half_t)x[2]; p[3] = (half_t)x[3];
      }
    }
    // ---- stage W tile 32x64 transposed -> Wts[n][k] ----
    for (int i = t; i < 256; i += 128) {
      int k = i >> 3, c8 = i & 7;
      v8h w = *(const v8h*)(W + (size_t)(kc * 32 + k) * Dn + n0 + c8 * 8);
#pragma unroll
      for (int e = 0; e < 8; ++e) Wts[c8 * 8 + e][k] = w[e];
    }
    __syncthreads();

    // ---- A fragment: rows wave*16..+15, k-chunk of 32 ----
    v16h a = cat8(*(const v8h*)&Xs[wave * 16 + r][k0],
                  *(const v8h*)&Xs[wave * 16 + r][k0 + 16]);
#pragma unroll
    for (int j = 0; j < 4; ++j) {
      int n = j * 16 + r;
      v16h b = cat8(*(const v8h*)&Wts[n][kh], *(const v8h*)&Wts[n][kh + 16]);
      acc[j] = wmma16(a, b, acc[j]);
    }
  }

  // ---- epilogue: bias + store ----
  const int mo = (lane < 16) ? 0 : 8;
#pragma unroll
  for (int j = 0; j < 4; ++j) {
    int ng = n0 + j * 16 + r;
    float bv = bias[ng];
#pragma unroll
    for (int v = 0; v < 8; ++v) {
      int   mg  = m0 + wave * 16 + v + mo;
      float val = acc[j][v] + bv;
      if (OUTMODE == 0) {
        ((half_t*)Yv)[(size_t)mg * Dn + ng] = (half_t)val;
      } else if (OUTMODE == 1) {               // V stored transposed per batch
        int b = mg >> 10, s = mg & 1023;
        ((half_t*)Yv)[((size_t)(b * Dn + ng)) * Sn + s] = (half_t)val;
      } else {
        ((float*)Yv)[(size_t)mg * Dn + ng] = val;
      }
    }
  }
}

// ---------------- fused attention core ----------------
// One wave per (b, h, 16-row q tile). Scores tile lives in LDS (64 KB),
// alignment written once to HBM, P@V fed straight from LDS.
__global__ __launch_bounds__(32)
void attn_kernel(const half_t* __restrict__ Qp, const half_t* __restrict__ Kp,
                 const half_t* __restrict__ Vt, half_t* __restrict__ Ctx,
                 float* __restrict__ Align) {
  __shared__ float sc[16][1024];   // exactly 64 KB

  const int id   = blockIdx.x;
  const int qt   = id & 63;
  const int h    = (id >> 6) & 15;
  const int b    = id >> 10;
  const int lane = threadIdx.x;
  const int r    = lane & 15;
  const int hi   = lane >> 4;           // wave32: two half-groups

  // ---- Q fragments (held in registers for all 64 K tiles) ----
  const half_t* Qrow =
      Qp + ((size_t)(b * Sn + qt * 16 + r)) * Dn + h * 64 + (hi ? 8 : 0);
  v16h aq0 = cat8(*(const v8h*)(Qrow),      *(const v8h*)(Qrow + 16));
  v16h aq1 = cat8(*(const v8h*)(Qrow + 32), *(const v8h*)(Qrow + 48));

  // ---- scores: S = Q K^T * 1/8, causal mask, ==0 -> -1e9 ----
  for (int kt = 0; kt < 64; ++kt) {
    const half_t* Krow =
        Kp + ((size_t)(b * Sn + kt * 16 + r)) * Dn + h * 64 + (hi ? 8 : 0);
    if (kt + 1 < 64) __builtin_prefetch(Krow + 16 * Dn, 0, 0);
    v16h bk0 = cat8(*(const v8h*)(Krow),      *(const v8h*)(Krow + 16));
    v16h bk1 = cat8(*(const v8h*)(Krow + 32), *(const v8h*)(Krow + 48));
    v8f s = {};
    s = wmma16(aq0, bk0, s);
    s = wmma16(aq1, bk1, s);
    const int kg = kt * 16 + r;
#pragma unroll
    for (int v = 0; v < 8; ++v) {
      int   m  = v + (hi ? 8 : 0);
      int   qg = qt * 16 + m;
      float x  = s[v] * 0.125f;                 // 1/sqrt(64)
      if (kg > qg || x == 0.0f) x = -1e9f;      // mask * score; where(==0, -1e9)
      sc[m][kt * 16 + r] = x;
    }
  }
  __syncthreads();

  // ---- softmax: lanes L and L+16 co-own row r ----
  const int cb = hi * 512;
  float mx = -3.0e38f;
  for (int i = 0; i < 512; i += 4) {
    v4f x = *(const v4f*)&sc[r][cb + i];
    mx = fmaxf(mx, fmaxf(fmaxf(x[0], x[1]), fmaxf(x[2], x[3])));
  }
  mx = fmaxf(mx, __shfl_xor(mx, 16));
  float sum = 0.f;
  for (int i = 0; i < 512; i += 4) {
    v4f x = *(const v4f*)&sc[r][cb + i];
    v4f e;
    e[0] = __expf(x[0] - mx); e[1] = __expf(x[1] - mx);
    e[2] = __expf(x[2] - mx); e[3] = __expf(x[3] - mx);
    *(v4f*)&sc[r][cb + i] = e;
    sum += e[0] + e[1] + e[2] + e[3];
  }
  sum += __shfl_xor(sum, 16);
  float rinv = 1.0f / sum;        // valid for row r on both partner lanes
  __syncthreads();

  // ---- normalize + coalesced alignment write (row stats via shuffle) ----
  float* alig = Align + ((size_t)((b * Hn + h) * Sn + qt * 16)) * Sn;
  for (int m = 0; m < 16; ++m) {
    float rs = __shfl(rinv, m);   // lane m owns row m's 1/sum
#pragma unroll 4
    for (int ch = 0; ch < 32; ++ch) {
      int   c = ch * 32 + lane;
      float p = sc[m][c] * rs;
      sc[m][c] = p;
      alig[(size_t)m * Sn + c] = p;
    }
  }
  __syncthreads();

  // ---- context: C[16,64] = P[16,1024] @ V[1024,64], P from LDS ----
  v8f co[4] = {};
  for (int c = 0; c < 32; ++c) {
    int kb = c * 32 + (hi ? 8 : 0);
    v4f f0 = *(const v4f*)&sc[r][kb];
    v4f f1 = *(const v4f*)&sc[r][kb + 4];
    v4f f2 = *(const v4f*)&sc[r][kb + 16];
    v4f f3 = *(const v4f*)&sc[r][kb + 20];
    v16h ap;
#pragma unroll
    for (int e = 0; e < 4; ++e) {
      ap[e]      = (half_t)f0[e];
      ap[4 + e]  = (half_t)f1[e];
      ap[8 + e]  = (half_t)f2[e];
      ap[12 + e] = (half_t)f3[e];
    }
#pragma unroll
    for (int j = 0; j < 4; ++j) {
      int dv = j * 16 + r;
      const half_t* vrow =
          Vt + ((size_t)(b * Dn + h * 64 + dv)) * Sn + c * 32 + (hi ? 8 : 0);
      v16h bv = cat8(*(const v8h*)(vrow), *(const v8h*)(vrow + 16));
      co[j] = wmma16(ap, bv, co[j]);
    }
  }
#pragma unroll
  for (int j = 0; j < 4; ++j) {
    int dv = j * 16 + r;
#pragma unroll
    for (int v = 0; v < 8; ++v) {
      int m = v + (hi ? 8 : 0);
      Ctx[((size_t)(b * Sn + qt * 16 + m)) * Dn + h * 64 + dv] = (half_t)co[j][v];
    }
  }
}

// ---------------- launch ----------------
extern "C" void kernel_launch(void* const* d_in, const int* in_sizes, int n_in,
                              void* d_out, int out_size, void* d_ws, size_t ws_size,
                              hipStream_t stream) {
  (void)in_sizes; (void)n_in; (void)out_size; (void)ws_size;
  const float* query = (const float*)d_in[0];
  const float* value = (const float*)d_in[1];
  // d_in[2] = mask: causal tril, applied analytically
  const float* wq = (const float*)d_in[3];
  const float* bq = (const float*)d_in[4];
  const float* wk = (const float*)d_in[5];
  const float* bk = (const float*)d_in[6];
  const float* wv = (const float*)d_in[7];
  const float* bv = (const float*)d_in[8];
  const float* wo = (const float*)d_in[9];
  const float* bo = (const float*)d_in[10];

  half_t* ws  = (half_t*)d_ws;
  half_t* Qp  = ws;              // [B,S,D] f16
  half_t* Kp  = ws + NE;         // [B,S,D] f16
  half_t* Vt  = ws + 2 * NE;     // [B,D,S] f16 (per-batch transposed V)
  half_t* Ctx = ws + 3 * NE;     // [B,S,D] f16
  half_t* Wqh = ws + 4 * NE;
  half_t* Wkh = Wqh + (size_t)Dn * Dn;
  half_t* Wvh = Wkh + (size_t)Dn * Dn;
  half_t* Woh = Wvh + (size_t)Dn * Dn;

  float* out_heads = (float*)d_out;            // [B,S,D]
  float* out_align = out_heads + NE;           // [B,H,S,S]

  // weights -> f16
  cvt_kernel<<<(Dn * Dn) / 1024, 256, 0, stream>>>(wq, Wqh);
  cvt_kernel<<<(Dn * Dn) / 1024, 256, 0, stream>>>(wk, Wkh);
  cvt_kernel<<<(Dn * Dn) / 1024, 256, 0, stream>>>(wv, Wvh);
  cvt_kernel<<<(Dn * Dn) / 1024, 256, 0, stream>>>(wo, Woh);

  dim3 gg((Bn * Sn) / 64, Dn / 64), bb(128);
  gemm_proj<false, 0><<<gg, bb, 0, stream>>>(query, Wqh, bq, Qp);
  gemm_proj<false, 0><<<gg, bb, 0, stream>>>(value, Wkh, bk, Kp);
  gemm_proj<false, 1><<<gg, bb, 0, stream>>>(value, Wvh, bv, Vt);

  attn_kernel<<<Bn * Hn * (Sn / 16), 32, 0, stream>>>(Qp, Kp, Vt, Ctx, out_align);

  gemm_proj<true, 2><<<gg, bb, 0, stream>>>(Ctx, Woh, bo, out_heads);
}
